// LLaMAModel_26070451486781
// MI455X (gfx1250) — matrix operand order repaired and checked by reading, not compiled, not run on hardware
//
#include <hip/hip_runtime.h>
#include <hip/hip_bf16.h>
#include <cstdint>

// ---------------------------------------------------------------------------
// LLaMA-2L forward for MI455X (gfx1250, wave32, WMMA).
// GEMMs: v_wmma_f32_16x16x32_f16 with double-buffered LDS tiles staged by
// CDNA5 async copies (global_load_async_to_lds_b128, ASYNCcnt-tracked).
// fp32 weights are converted to f16 in flight (reg-staged loads overlapped
// with the WMMA burst); f16 operands use the pure async DMA path.
// ---------------------------------------------------------------------------

#define SEQ   1024
#define DMODEL 2048
#define NHEAD 16
#define HEADD 128
#define NLAYER 2
#define NPREF 64
#define NVOCAB 32000
#define DHID  5632

typedef _Float16 h8  __attribute__((ext_vector_type(8)));
typedef _Float16 h16 __attribute__((ext_vector_type(16)));
typedef float    f8  __attribute__((ext_vector_type(8)));
typedef float    f4v __attribute__((ext_vector_type(4)));

// CDNA5 async global->LDS copy (16B per lane). VDST = LDS byte address,
// VADDR = 64-bit global address, no SADDR. Tracked by ASYNCcnt.
__device__ __forceinline__ void async_copy_b128(uint32_t lds_off, unsigned long long gaddr)
{
    asm volatile("global_load_async_to_lds_b128 %0, %1, off"
                 :: "v"(lds_off), "v"(gaddr) : "memory");
}
__device__ __forceinline__ void async_wait0()
{
    asm volatile("s_wait_asynccnt 0x0" ::: "memory");
}

// ---------------------------------------------------------------------------
// WMMA GEMM:  C[M,N] (f32) (+)= alpha * A[M,K](f16) * B[N,K]^T (TB = f32|f16)
// Block: 256 threads = 8 waves, tile 128(M) x 64(N), K-step 32.
// Wave (wm,wn) grid = 2 x 4; each wave computes a 64x16 slab = 4 wmma accs.
// Double-buffered LDS; next tile staged asynchronously during compute.
// ---------------------------------------------------------------------------
#define APITCH 40   // halves per LDS row (pad: 80B row stride, 16B aligned)

template <typename TB, bool ACC>
__global__ __launch_bounds__(256) void gemm_wmma(
    float* __restrict__ C, const _Float16* __restrict__ A, const TB* __restrict__ B,
    int M, int N, int K, int lda, int ldb, int ldc,
    long bsA, long bsB, long bsC, const float* __restrict__ alphaPerBatch)
{
    __shared__ _Float16 sA[2][128 * APITCH];
    __shared__ _Float16 sB[2][64 * APITCH];

    const int tid  = threadIdx.x;
    const int lane = tid & 31;
    const int wv   = tid >> 5;
    const int wm   = (wv >> 2) * 64;   // 0 or 64
    const int wn   = (wv & 3) * 16;    // 0..48

    const long zb = blockIdx.z;
    const _Float16* Ab = A + zb * bsA + (long)blockIdx.y * 128 * lda;
    const TB*       Bb = B + zb * bsB + (long)blockIdx.x * 64 * ldb;
    float*          Cb = C + zb * bsC + (long)blockIdx.y * 128 * ldc + (long)blockIdx.x * 64;
    const float alpha = alphaPerBatch ? alphaPerBatch[zb] : 1.0f;

    // A staging: thread -> (row = tid/2, 16 halves at (tid&1)*16) : 2 x b128
    const int arow = tid >> 1;
    const int asel = (tid & 1) * 16;
    // B staging: thread -> (row = tid/4, 8 elems at (tid&3)*8)    : 1 x b128 (f16)
    const int brow = tid >> 2;
    const int bsel = (tid & 3) * 8;

    auto stageA = [&](int b, int k0) {
        const _Float16* ag = Ab + (long)arow * lda + k0 + asel;
        const uint32_t l0 = (uint32_t)(uintptr_t)&sA[b][arow * APITCH + asel];
        async_copy_b128(l0,      (unsigned long long)(uintptr_t)ag);
        async_copy_b128(l0 + 16, (unsigned long long)(uintptr_t)(ag + 8));
    };
    auto stageB16 = [&](int b, int k0) {
        const _Float16* bg = (const _Float16*)Bb + (long)brow * ldb + k0 + bsel;
        async_copy_b128((uint32_t)(uintptr_t)&sB[b][brow * APITCH + bsel],
                        (unsigned long long)(uintptr_t)bg);
    };
    auto loadB32 = [&](int k0, f4v& r0, f4v& r1) {
        const float* bg = (const float*)Bb + (long)brow * ldb + k0 + bsel;
        r0 = *(const f4v*)bg;
        r1 = *(const f4v*)(bg + 4);
        __builtin_prefetch(bg + 64, 0, 1);     // global_prefetch_b8, tile k+2
    };
    auto commitB32 = [&](int b, const f4v& r0, const f4v& r1) {
        h8 hv;
        hv[0] = (_Float16)r0[0]; hv[1] = (_Float16)r0[1];
        hv[2] = (_Float16)r0[2]; hv[3] = (_Float16)r0[3];
        hv[4] = (_Float16)r1[0]; hv[5] = (_Float16)r1[1];
        hv[6] = (_Float16)r1[2]; hv[7] = (_Float16)r1[3];
        *(h8*)&sB[b][brow * APITCH + bsel] = hv;
    };

    f8 acc[4] = {f8{}, f8{}, f8{}, f8{}};

    // ---- prologue: stage tile 0 ----
    stageA(0, 0);
    if constexpr (sizeof(TB) == 4) {
        f4v r0, r1;
        loadB32(0, r0, r1);
        commitB32(0, r0, r1);
    } else {
        stageB16(0, 0);
    }
    async_wait0();
    __syncthreads();

    int nbuf = 0;
    for (int k0 = 0; k0 < K; k0 += 32) {
        const int buf = nbuf;
        nbuf ^= 1;
        const bool hasNext = (k0 + 32 < K);

        // ---- issue next tile's loads (async DMA / reg-staged) ----
        f4v r0, r1;
        if (hasNext) {
            stageA(nbuf, k0 + 32);
            if constexpr (sizeof(TB) == 4) loadB32(k0 + 32, r0, r1);
            else                           stageB16(nbuf, k0 + 32);
        }

        // ---- compute on current tile ----
        {
            const int bn = wn + (lane & 15);
            const int bk = (lane >> 4) * 16;      // lanes 0-15: K 0..15, 16-31: K 16..31
            h8 b0 = *(const h8*)&sB[buf][bn * APITCH + bk];
            h8 b1 = *(const h8*)&sB[buf][bn * APITCH + bk + 8];
            h16 bf = __builtin_shufflevector(b0, b1, 0,1,2,3,4,5,6,7,8,9,10,11,12,13,14,15);

            const int am = lane & 15;
            const int ks = (lane >> 4) * 16;      // lanes 0-15: K 0..15, 16-31: K 16..31
#pragma unroll
            for (int t = 0; t < 4; ++t) {
                const int ar = wm + t * 16 + am;
                h8 a0 = *(const h8*)&sA[buf][ar * APITCH + ks];
                h8 a1 = *(const h8*)&sA[buf][ar * APITCH + ks + 8];
                h16 af = __builtin_shufflevector(a0, a1, 0,1,2,3,4,5,6,7,8,9,10,11,12,13,14,15);
                acc[t] = __builtin_amdgcn_wmma_f32_16x16x32_f16(
                    false, af, false, bf, (short)0, acc[t], false, false);
            }
        }

        // ---- retire next tile's staging, fence, swap ----
        if (hasNext) {
            if constexpr (sizeof(TB) == 4) commitB32(nbuf, r0, r1);
            async_wait0();
            __syncthreads();
        }
    }

    // ---- store: C/D layout -> lanes 0-15: M r, lanes 16-31: M r+8; N = lane&15 ----
    const int ml = (lane >> 4) * 8;
    const int nl = lane & 15;
#pragma unroll
    for (int t = 0; t < 4; ++t) {
#pragma unroll
        for (int r = 0; r < 8; ++r) {
            const long m = wm + t * 16 + r + ml;
            float* cp = Cb + m * (long)ldc + wn + nl;
            if constexpr (ACC) *cp += alpha * acc[t][r];
            else               *cp  = alpha * acc[t][r];
        }
    }
}

// ---------------------------------------------------------------------------
// Elementwise / reduction helper kernels
// ---------------------------------------------------------------------------

__global__ __launch_bounds__(256) void embed_kernel(
    const int* __restrict__ ids, const float* __restrict__ embed, float* __restrict__ h)
{
    const int s = blockIdx.x;
    const long row = (long)ids[s] * DMODEL;
#pragma unroll
    for (int i = 0; i < DMODEL / 256; ++i) {
        const int d = threadIdx.x + i * 256;
        h[(long)s * DMODEL + d] = embed[row + d];
    }
}

__global__ __launch_bounds__(256) void rmsnorm_kernel(
    const float* __restrict__ x, const float* __restrict__ w, _Float16* __restrict__ y16)
{
    __shared__ float red[256];
    const int s = blockIdx.x;
    const float* row = x + (long)s * DMODEL;
    float ss = 0.0f;
#pragma unroll
    for (int i = 0; i < DMODEL / 256; ++i) {
        const float v = row[threadIdx.x + i * 256];
        ss += v * v;
    }
    red[threadIdx.x] = ss;
    __syncthreads();
    for (int off = 128; off > 0; off >>= 1) {
        if (threadIdx.x < off) red[threadIdx.x] += red[threadIdx.x + off];
        __syncthreads();
    }
    const float rstd = rsqrtf(red[0] / (float)DMODEL + 1e-6f);
#pragma unroll
    for (int i = 0; i < DMODEL / 256; ++i) {
        const int d = threadIdx.x + i * 256;
        y16[(long)s * DMODEL + d] = (_Float16)(row[d] * rstd * w[d]);
    }
}

// RoPE + f32->f16: q' = q*cos + rotate_half(q)*sin, per head of 128.
__global__ __launch_bounds__(256) void rope_kernel(
    const float* __restrict__ q, const float* __restrict__ cs, const float* __restrict__ sn,
    _Float16* __restrict__ out16)
{
    const int s = blockIdx.y;
    const int d = blockIdx.x * 256 + threadIdx.x;
    const int hd = d & (HEADD - 1);
    const float x = q[(long)s * DMODEL + d];
    const float r = (hd < HEADD / 2) ? -q[(long)s * DMODEL + d + HEADD / 2]
                                     :  q[(long)s * DMODEL + d - HEADD / 2];
    out16[(long)s * DMODEL + d] =
        (_Float16)(x * cs[(long)s * HEADD + hd] + r * sn[(long)s * HEADD + hd]);
}

// V [S, H*HD] f32 -> vT [H][HD][S] f16 (rows of length S, K-major for A*B^T)
__global__ __launch_bounds__(256) void transpose_v_kernel(
    const float* __restrict__ v, _Float16* __restrict__ vT)
{
    const int s  = blockIdx.x * 256 + threadIdx.x;
    const int hd = blockIdx.y;
    const int h  = blockIdx.z;
    vT[((long)h * HEADD + hd) * SEQ + s] =
        (_Float16)v[(long)s * DMODEL + h * HEADD + hd];
}

// prefix_v[l] [H,P,HD] f32 -> pvT [H][HD][P] f16
__global__ __launch_bounds__(256) void transpose_pv_kernel(
    const float* __restrict__ pv, _Float16* __restrict__ pvT)
{
    const int idx = blockIdx.x * 256 + threadIdx.x;   // H*HD*P = 131072
    const int p   = idx & (NPREF - 1);
    const int hd  = (idx >> 6) & (HEADD - 1);
    const int h   = idx >> 13;
    pvT[((long)h * HEADD + hd) * NPREF + p] =
        (_Float16)pv[((long)h * NPREF + p) * HEADD + hd];
}

// softmax over row of S with causal mask; out f16. grid (S, H), 256 thr.
__global__ __launch_bounds__(256) void softmax_mask_kernel(
    const float* __restrict__ sc, const float* __restrict__ mask,
    _Float16* __restrict__ out, float scale)
{
    __shared__ float red[256];
    const int s = blockIdx.x, h = blockIdx.y;
    const float* row  = sc   + ((long)h * SEQ + s) * SEQ;
    const float* mrow = mask + (long)s * SEQ;
    float v[SEQ / 256];
    float mx = -3.4e38f;
#pragma unroll
    for (int i = 0; i < SEQ / 256; ++i) {
        const int j = threadIdx.x + i * 256;
        v[i] = row[j] * scale + mrow[j];
        mx = fmaxf(mx, v[i]);
    }
    red[threadIdx.x] = mx;
    __syncthreads();
    for (int off = 128; off > 0; off >>= 1) {
        if (threadIdx.x < off) red[threadIdx.x] = fmaxf(red[threadIdx.x], red[threadIdx.x + off]);
        __syncthreads();
    }
    mx = red[0];
    __syncthreads();
    float sum = 0.0f;
#pragma unroll
    for (int i = 0; i < SEQ / 256; ++i) { v[i] = expf(v[i] - mx); sum += v[i]; }
    red[threadIdx.x] = sum;
    __syncthreads();
    for (int off = 128; off > 0; off >>= 1) {
        if (threadIdx.x < off) red[threadIdx.x] += red[threadIdx.x + off];
        __syncthreads();
    }
    const float inv = 1.0f / red[0];
    _Float16* orow = out + ((long)h * SEQ + s) * SEQ;
#pragma unroll
    for (int i = 0; i < SEQ / 256; ++i)
        orow[threadIdx.x + i * 256] = (_Float16)(v[i] * inv);
}

// prefix softmax over P=64, no mask. grid (S, H), 64 thr.
__global__ __launch_bounds__(64) void psoftmax_kernel(
    const float* __restrict__ sc, _Float16* __restrict__ out, float scale)
{
    __shared__ float red[64];
    const int s = blockIdx.x, h = blockIdx.y;
    const float* row = sc + ((long)h * SEQ + s) * NPREF;
    float v = row[threadIdx.x] * scale;
    red[threadIdx.x] = v;
    __syncthreads();
    for (int off = 32; off > 0; off >>= 1) {
        if (threadIdx.x < off) red[threadIdx.x] = fmaxf(red[threadIdx.x], red[threadIdx.x + off]);
        __syncthreads();
    }
    const float mx = red[0];
    __syncthreads();
    v = expf(v - mx);
    red[threadIdx.x] = v;
    __syncthreads();
    for (int off = 32; off > 0; off >>= 1) {
        if (threadIdx.x < off) red[threadIdx.x] += red[threadIdx.x + off];
        __syncthreads();
    }
    out[((long)h * SEQ + s) * NPREF + threadIdx.x] = (_Float16)(v / red[0]);
}

__global__ __launch_bounds__(256) void cvt16_kernel(
    const float* __restrict__ x, _Float16* __restrict__ y, long n)
{
    const long i = (long)blockIdx.x * 256 + threadIdx.x;
    if (i < n) y[i] = (_Float16)x[i];
}

__global__ __launch_bounds__(256) void silu_mul_kernel(
    const float* __restrict__ g, const float* __restrict__ u, _Float16* __restrict__ out)
{
    const long i = (long)blockIdx.x * 256 + threadIdx.x;
    const float gv = g[i];
    out[i] = (_Float16)((gv / (1.0f + expf(-gv))) * u[i]);
}

// ---------------------------------------------------------------------------
// Orchestration
// ---------------------------------------------------------------------------
extern "C" void kernel_launch(void* const* d_in, const int* in_sizes, int n_in,
                              void* d_out, int out_size, void* d_ws, size_t ws_size,
                              hipStream_t stream)
{
    (void)in_sizes; (void)n_in; (void)out_size; (void)ws_size;

    const int*   ids     = (const int*)  d_in[0];
    const float* mask    = (const float*)d_in[1];
    const float* cs      = (const float*)d_in[2];
    const float* sn      = (const float*)d_in[3];
    const float* embed   = (const float*)d_in[4];
    const float* wq      = (const float*)d_in[5];
    const float* wk      = (const float*)d_in[6];
    const float* wv      = (const float*)d_in[7];
    const float* wo      = (const float*)d_in[8];
    const float* ln1     = (const float*)d_in[9];
    const float* ln2     = (const float*)d_in[10];
    const float* w_gate  = (const float*)d_in[11];
    const float* w_up    = (const float*)d_in[12];
    const float* w_down  = (const float*)d_in[13];
    const float* pk      = (const float*)d_in[14];
    const float* pv      = (const float*)d_in[15];
    const float* gates   = (const float*)d_in[16];
    const float* fln     = (const float*)d_in[17];
    const float* lm_head = (const float*)d_in[18];
    float* logits = (float*)d_out;

    // ---- workspace carve (aligned to 256B) ----
    uint8_t* p = (uint8_t*)d_ws;
    size_t off = 0;
    auto alloc = [&](size_t bytes) {
        void* r = p + off;
        off += (bytes + 255) & ~(size_t)255;
        return r;
    };
    float*    hbuf  = (float*)   alloc((size_t)SEQ * DMODEL * 4);
    _Float16* xn16  = (_Float16*)alloc((size_t)SEQ * DMODEL * 2);
    float*    qf    = (float*)   alloc((size_t)SEQ * DMODEL * 4);
    float*    kf    = (float*)   alloc((size_t)SEQ * DMODEL * 4);
    float*    vf    = (float*)   alloc((size_t)SEQ * DMODEL * 4);
    _Float16* q16   = (_Float16*)alloc((size_t)SEQ * DMODEL * 2);
    _Float16* k16   = (_Float16*)alloc((size_t)SEQ * DMODEL * 2);
    _Float16* vT16  = (_Float16*)alloc((size_t)NHEAD * HEADD * SEQ * 2);
    float*    scbuf = (float*)   alloc((size_t)NHEAD * SEQ * SEQ * 4);
    _Float16* sm16  = (_Float16*)alloc((size_t)NHEAD * SEQ * SEQ * 2);
    float*    psc   = (float*)   alloc((size_t)NHEAD * SEQ * NPREF * 4);
    _Float16* psm16 = (_Float16*)alloc((size_t)NHEAD * SEQ * NPREF * 2);
    float*    attnO = (float*)   alloc((size_t)SEQ * DMODEL * 4);
    _Float16* at16  = (_Float16*)alloc((size_t)SEQ * DMODEL * 2);
    _Float16* pvT16 = (_Float16*)alloc((size_t)NHEAD * HEADD * NPREF * 2);
    float*    gbuf  = (float*)   alloc((size_t)SEQ * DHID * 4);
    float*    ubuf  = (float*)   alloc((size_t)SEQ * DHID * 4);
    _Float16* gu16  = (_Float16*)alloc((size_t)SEQ * DHID * 2);

    const float scale = 0.08838834764831845f;   // 1/sqrt(128)
    const dim3 blk(256);

    auto gemm_grid = [](int M, int N, int Z) { return dim3(N / 64, M / 128, Z); };

    // ---- embedding ----
    embed_kernel<<<dim3(SEQ), blk, 0, stream>>>(ids, embed, hbuf);

    for (int l = 0; l < NLAYER; ++l) {
        const long wOff  = (long)l * DMODEL * DMODEL;
        const long gOff  = (long)l * DHID * DMODEL;

        // 1) hn = rmsnorm(h, ln1) -> f16
        rmsnorm_kernel<<<dim3(SEQ), blk, 0, stream>>>(hbuf, ln1 + (long)l * DMODEL, xn16);

        // 2) Q/K/V projections (N=D, K=D)
        gemm_wmma<float, false><<<gemm_grid(SEQ, DMODEL, 1), blk, 0, stream>>>(
            qf, xn16, wq + wOff, SEQ, DMODEL, DMODEL, DMODEL, DMODEL, DMODEL, 0, 0, 0, nullptr);
        gemm_wmma<float, false><<<gemm_grid(SEQ, DMODEL, 1), blk, 0, stream>>>(
            kf, xn16, wk + wOff, SEQ, DMODEL, DMODEL, DMODEL, DMODEL, DMODEL, 0, 0, 0, nullptr);
        gemm_wmma<float, false><<<gemm_grid(SEQ, DMODEL, 1), blk, 0, stream>>>(
            vf, xn16, wv + wOff, SEQ, DMODEL, DMODEL, DMODEL, DMODEL, DMODEL, 0, 0, 0, nullptr);

        // 3) RoPE -> f16; V transpose -> f16
        rope_kernel<<<dim3(DMODEL / 256, SEQ), blk, 0, stream>>>(qf, cs, sn, q16);
        rope_kernel<<<dim3(DMODEL / 256, SEQ), blk, 0, stream>>>(kf, cs, sn, k16);
        transpose_v_kernel<<<dim3(SEQ / 256, HEADD, NHEAD), blk, 0, stream>>>(vf, vT16);
        transpose_pv_kernel<<<dim3(NHEAD * HEADD * NPREF / 256), blk, 0, stream>>>(
            pv + (long)l * NHEAD * NPREF * HEADD, pvT16);

        // 4) scores = Q K^T per head (batched over z = head)
        gemm_wmma<_Float16, false><<<gemm_grid(SEQ, SEQ, NHEAD), blk, 0, stream>>>(
            scbuf, q16, k16, SEQ, SEQ, HEADD, DMODEL, DMODEL, SEQ,
            HEADD, HEADD, (long)SEQ * SEQ, nullptr);
        softmax_mask_kernel<<<dim3(SEQ, NHEAD), blk, 0, stream>>>(scbuf, mask, sm16, scale);

        // 5) prefix scores = Q pk^T (B f32, N=P, K=HD)
        gemm_wmma<float, false><<<gemm_grid(SEQ, NPREF, NHEAD), blk, 0, stream>>>(
            psc, q16, pk + (long)l * NHEAD * NPREF * HEADD, SEQ, NPREF, HEADD,
            DMODEL, HEADD, NPREF, HEADD, (long)NPREF * HEADD, (long)SEQ * NPREF, nullptr);
        psoftmax_kernel<<<dim3(SEQ, NHEAD), dim3(64), 0, stream>>>(psc, psm16, scale);

        // 6) attn = softmax(scores) * V  -> [s][h*HD+hd]
        gemm_wmma<_Float16, false><<<gemm_grid(SEQ, HEADD, NHEAD), blk, 0, stream>>>(
            attnO, sm16, vT16, SEQ, HEADD, SEQ, SEQ, SEQ, DMODEL,
            (long)SEQ * SEQ, (long)HEADD * SEQ, HEADD, nullptr);

        // 7) attn += gate[h] * (psm * prefix_v)
        gemm_wmma<_Float16, true><<<gemm_grid(SEQ, HEADD, NHEAD), blk, 0, stream>>>(
            attnO, psm16, pvT16, SEQ, HEADD, NPREF, NPREF, NPREF, DMODEL,
            (long)SEQ * NPREF, (long)HEADD * NPREF, HEADD, gates + (long)l * NHEAD);

        // 8) h += attn * wo^T
        cvt16_kernel<<<dim3((SEQ * DMODEL) / 256), blk, 0, stream>>>(attnO, at16, (long)SEQ * DMODEL);
        gemm_wmma<float, true><<<gemm_grid(SEQ, DMODEL, 1), blk, 0, stream>>>(
            hbuf, at16, wo + wOff, SEQ, DMODEL, DMODEL, DMODEL, DMODEL, DMODEL, 0, 0, 0, nullptr);

        // 9) SwiGLU MLP
        rmsnorm_kernel<<<dim3(SEQ), blk, 0, stream>>>(hbuf, ln2 + (long)l * DMODEL, xn16);
        gemm_wmma<float, false><<<gemm_grid(SEQ, DHID, 1), blk, 0, stream>>>(
            gbuf, xn16, w_gate + gOff, SEQ, DHID, DMODEL, DMODEL, DMODEL, DHID, 0, 0, 0, nullptr);
        gemm_wmma<float, false><<<gemm_grid(SEQ, DHID, 1), blk, 0, stream>>>(
            ubuf, xn16, w_up + gOff, SEQ, DHID, DMODEL, DMODEL, DMODEL, DHID, 0, 0, 0, nullptr);
        silu_mul_kernel<<<dim3((SEQ * DHID) / 256), blk, 0, stream>>>(gbuf, ubuf, gu16);
        gemm_wmma<float, true><<<gemm_grid(SEQ, DMODEL, 1), blk, 0, stream>>>(
            hbuf, gu16, w_down + (long)l * DMODEL * DHID, SEQ, DMODEL, DHID,
            DHID, DHID, DMODEL, 0, 0, 0, nullptr);
    }

    // ---- final norm + LM head (M=1024, N=32000, K=2048) ----
    rmsnorm_kernel<<<dim3(SEQ), blk, 0, stream>>>(hbuf, fln, xn16);
    gemm_wmma<float, false><<<gemm_grid(SEQ, NVOCAB, 1), blk, 0, stream>>>(
        logits, xn16, lm_head, SEQ, NVOCAB, DMODEL, DMODEL, DMODEL, NVOCAB, 0, 0, 0, nullptr);
}
